// Decoder_70781061038948
// MI455X (gfx1250) — matrix operand order repaired and checked by reading, not compiled, not run on hardware
//
#include <hip/hip_runtime.h>

// ---------------------------------------------------------------------------
// Types
// ---------------------------------------------------------------------------
typedef __bf16 bf16;
typedef __attribute__((ext_vector_type(16))) __bf16 v16bf;
typedef __attribute__((ext_vector_type(8)))  float  v8f;
typedef __attribute__((ext_vector_type(4)))  unsigned int u32x4;

union FragBF {            // 32 bytes: one WMMA 16-bit operand fragment (8 VGPRs)
    v16bf v;
    u32x4 q[2];
};

__device__ __forceinline__ bf16 f2bf(float f) {
    union { float f; unsigned u; } a; a.f = f;
    unsigned r = a.u + 0x7FFFu + ((a.u >> 16) & 1u);   // round-to-nearest-even
    union { unsigned short s; bf16 b; } o; o.s = (unsigned short)(r >> 16);
    return o.b;
}

// CDNA5 async global->LDS copy (ASYNCcnt-tracked, no VGPR staging).
// lds_off: wave-relative LDS byte offset (low 32 bits of flat shared address).
__device__ __forceinline__ void async_load_b128(unsigned lds_off, const void* gaddr) {
    asm volatile("global_load_async_to_lds_b128 %0, %1, off"
                 :: "v"(lds_off), "v"(gaddr) : "memory");
}
__device__ __forceinline__ void wait_asynccnt0() {
#if __has_builtin(__builtin_amdgcn_s_wait_asynccnt)
    __builtin_amdgcn_s_wait_asynccnt(0);
#else
    asm volatile("s_wait_asynccnt 0" ::: "memory");
#endif
}

// ---------------------------------------------------------------------------
// Problem constants
// ---------------------------------------------------------------------------
#define BATCH   8192
#define POLYK   10          // polynomial degree
#define IN1     20          // 2*K roots (re/im interleaved)
#define HID     1024
#define OUTN    1024        // 2**K
#define NB      8           // spline bases per input
#define K1PAD   192         // 20*9 = 180 -> padded to 6*32
#define K2      (HID * 9)   // 9216 = 288*32, no pad needed

// ---------------------------------------------------------------------------
// 1. Polynomial roots: Durand-Kerner on monic p, one thread per polynomial
// ---------------------------------------------------------------------------
struct cplx { float re, im; };
__device__ __forceinline__ cplx cmul(cplx a, cplx b) {
    return { a.re * b.re - a.im * b.im, a.re * b.im + a.im * b.re };
}
__device__ __forceinline__ cplx cadd(cplx a, cplx b) { return { a.re + b.re, a.im + b.im }; }
__device__ __forceinline__ cplx csub(cplx a, cplx b) { return { a.re - b.re, a.im - b.im }; }
__device__ __forceinline__ cplx cdiv(cplx a, cplx b) {
    float d = b.re * b.re + b.im * b.im;
    d = fmaxf(d, 1e-30f);
    float inv = 1.0f / d;
    return { (a.re * b.re + a.im * b.im) * inv, (a.im * b.re - a.re * b.im) * inv };
}

__global__ __launch_bounds__(256)
void dk_roots_kernel(const float* __restrict__ x, float* __restrict__ roots, int nbatch) {
    int b = blockIdx.x * blockDim.x + threadIdx.x;
    if (b >= nbatch) return;

    const float* p = x + (size_t)b * (POLYK + 1) * 2;
    cplx c0 = { p[0], p[1] };
    float d0 = fmaxf(c0.re * c0.re + c0.im * c0.im, 1e-30f);
    float inv0 = 1.0f / d0;
    cplx ic0 = { c0.re * inv0, -c0.im * inv0 };

    cplx a[POLYK];                       // monic coeffs a1..a10
#pragma unroll
    for (int j = 0; j < POLYK; ++j) {
        cplx cj = { p[2 * (j + 1)], p[2 * (j + 1) + 1] };
        a[j] = cmul(cj, ic0);
    }

    // init z_i = (0.4 + 0.9i)^(i+1)
    cplx z[POLYK];
    cplx w = { 0.4f, 0.9f }, zz = w;
#pragma unroll
    for (int i = 0; i < POLYK; ++i) { z[i] = zz; zz = cmul(zz, w); }

    for (int it = 0; it < 64; ++it) {
#pragma unroll
        for (int i = 0; i < POLYK; ++i) {
            cplx v = { 1.0f, 0.0f };
#pragma unroll
            for (int j = 0; j < POLYK; ++j) v = cadd(cmul(v, z[i]), a[j]);
            cplx den = { 1.0f, 0.0f };
#pragma unroll
            for (int j = 0; j < POLYK; ++j)
                if (j != i) den = cmul(den, csub(z[i], z[j]));
            cplx corr = cdiv(v, den);
            z[i] = csub(z[i], corr);
        }
    }

    float* o = roots + (size_t)b * IN1;
#pragma unroll
    for (int i = 0; i < POLYK; ++i) { o[2 * i] = z[i].re; o[2 * i + 1] = z[i].im; }
}

// ---------------------------------------------------------------------------
// 2. B-spline (order 3, grid_size 5, uniform grid h=0.4 on [-2.2, 2.2])
// ---------------------------------------------------------------------------
__device__ __forceinline__ void bspline8(float x, float out[NB]) {
    const float h = 0.4f;
    float g[12];
#pragma unroll
    for (int j = 0; j < 12; ++j) g[j] = 0.4f * (float)j - 2.2f;
    float bs[11];
#pragma unroll
    for (int j = 0; j < 11; ++j) bs[j] = (x >= g[j] && x < g[j + 1]) ? 1.0f : 0.0f;
#pragma unroll
    for (int k = 1; k <= 3; ++k) {
        float inv = 1.0f / ((float)k * h);
#pragma unroll
        for (int j = 0; j < 11 - k; ++j)
            bs[j] = ((x - g[j]) * bs[j] + (g[j + k + 1] - x) * bs[j + 1]) * inv;
    }
#pragma unroll
    for (int j = 0; j < NB; ++j) out[j] = bs[j];
}

// Expand activations X[B][IN] -> A[B][Kpad] bf16, 9 channels per input:
// channel 0 = silu(x), channels 1..8 = b-spline bases. Pad tail zeroed.
__global__ __launch_bounds__(256)
void expand_kernel(const float* __restrict__ X, bf16* __restrict__ A,
                   int nbatch, int nin, int kpad) {
    int t = blockIdx.x * blockDim.x + threadIdx.x;
    if (t >= nbatch * nin) return;
    int b = t / nin, i = t - b * nin;
    float x = X[(size_t)b * nin + i];
    float s = x / (1.0f + __expf(-x));       // silu
    float bs[NB];
    bspline8(x, bs);
    bf16* p = A + (size_t)b * kpad + (size_t)i * 9;
    p[0] = f2bf(s);
#pragma unroll
    for (int c = 0; c < NB; ++c) p[1 + c] = f2bf(bs[c]);
    if (i == 0) {
        for (int k = nin * 9; k < kpad; ++k) A[(size_t)b * kpad + k] = f2bf(0.0f);
    }
}

// Pack weights into Wt[N][Kpad] bf16 (transposed: K contiguous per output row):
// Wt[n][i*9+0] = bw[n][i]; Wt[n][i*9+1+s] = sw[n][i][s] * sc[n][i]
__global__ __launch_bounds__(256)
void pack_w_kernel(const float* __restrict__ bw, const float* __restrict__ sw,
                   const float* __restrict__ sc, bf16* __restrict__ Wt,
                   int nout, int nin, int kpad) {
    int t = blockIdx.x * blockDim.x + threadIdx.x;
    if (t >= nout * nin) return;
    int n = t / nin, i = t - n * nin;
    float scale = sc[(size_t)n * nin + i];
    bf16* p = Wt + (size_t)n * kpad + (size_t)i * 9;
    p[0] = f2bf(bw[(size_t)n * nin + i]);
    const float* s = sw + ((size_t)n * nin + i) * NB;
#pragma unroll
    for (int c = 0; c < NB; ++c) p[1 + c] = f2bf(s[c] * scale);
    if (i == 0) {
        for (int k = nin * 9; k < kpad; ++k) Wt[(size_t)n * kpad + k] = f2bf(0.0f);
    }
}

// ---------------------------------------------------------------------------
// 3. bf16 WMMA GEMM:  C[M][N] = A[M][K] * Wt[N][K]^T   (f32 accumulate)
//    block = 256 threads (8 waves), tile 128x128, K stepped by 32.
//    LDS double-buffered; next tile staged with GLOBAL_LOAD_ASYNC_TO_LDS_B128
//    (ASYNCcnt) so the copy bypasses VGPRs and overlaps the WMMA phase.
//    Wave w: (w&3) -> 32-row group, (w>>2) -> 64-col group;
//    each wave owns a 2x4 grid of 16x16 WMMA accumulators (8 WMMA/chunk).
// ---------------------------------------------------------------------------
#define TILE_M 128
#define TILE_N 128
#define LDSS   40                    // LDS row stride in bf16 (80B: conflict-free, 16B aligned)
#define ABUF   (TILE_M * LDSS)       // one A buffer, bf16 elements
#define BBUF   (TILE_N * LDSS)       // one B buffer, bf16 elements

__global__ __launch_bounds__(256)
void kan_gemm_bf16(const bf16* __restrict__ A, const bf16* __restrict__ Wt,
                   float* __restrict__ C, int M, int N, int K) {
    __shared__ __align__(16) bf16 sA[2 * ABUF];
    __shared__ __align__(16) bf16 sB[2 * BBUF];

    const int tid   = threadIdx.x;
    const int lane  = tid & 31;
    const int wave  = tid >> 5;
    const int wm    = wave & 3;          // 32-row group within tile
    const int wn    = wave >> 2;         // 64-col group within tile
    const int m0    = blockIdx.y * TILE_M;
    const int n0    = blockIdx.x * TILE_N;
    const int lrow  = lane & 15;
    const int khalf = lane >> 4;         // upper/lower K sub-block per ISA layout

    // global->LDS mapping: A and B each need 128x32 bf16 per chunk
    // = 512 chunks of 8 bf16 -> 2 chunks per thread per operand
    const int grow = tid >> 2;           // 0..63
    const int gkc  = (tid & 3) * 8;      // 0,8,16,24

    const bf16* gA0 = A  + (size_t)(m0 + grow)      * K + gkc;
    const bf16* gA1 = A  + (size_t)(m0 + grow + 64) * K + gkc;
    const bf16* gB0 = Wt + (size_t)(n0 + grow)      * K + gkc;
    const bf16* gB1 = Wt + (size_t)(n0 + grow + 64) * K + gkc;

    // wave-relative LDS byte offsets (flat shared address truncates to offset)
    const unsigned ldsA = (unsigned)(uintptr_t)(sA);
    const unsigned ldsB = (unsigned)(uintptr_t)(sB);
    const unsigned o0   = (unsigned)((grow * LDSS + gkc) * 2);
    const unsigned o1   = (unsigned)(((grow + 64) * LDSS + gkc) * 2);

    const int nk = K >> 5;

    // prologue: async-stage chunk 0 into buffer 0
    async_load_b128(ldsA + o0, gA0);
    async_load_b128(ldsA + o1, gA1);
    async_load_b128(ldsB + o0, gB0);
    async_load_b128(ldsB + o1, gB1);
    wait_asynccnt0();
    __syncthreads();

    v8f acc[2][4] = {};

    for (int i = 0; i < nk; ++i) {
        const int cur = i & 1;
        if (i + 1 < nk) {                // async-stage next chunk into other buffer
            const unsigned bA = ldsA + (unsigned)((cur ^ 1) * ABUF * 2);
            const unsigned bB = ldsB + (unsigned)((cur ^ 1) * BBUF * 2);
            const size_t ko = (size_t)(i + 1) << 5;
            async_load_b128(bA + o0, gA0 + ko);
            async_load_b128(bA + o1, gA1 + ko);
            async_load_b128(bB + o0, gB0 + ko);
            async_load_b128(bB + o1, gB1 + ko);
        }

        const bf16* cA = sA + cur * ABUF;
        const bf16* cB = sB + cur * BBUF;

        FragBF af[2], bfr[4];
#pragma unroll
        for (int r = 0; r < 2; ++r) {
            // A 16x32 bf16 layout: lane<16 -> K{0..7,16..23}; lane>=16 -> K{8..15,24..31}
            const bf16* pa = cA + (wm * 32 + r * 16 + lrow) * LDSS;
            af[r].q[0] = *(const u32x4*)(pa + khalf * 8);
            af[r].q[1] = *(const u32x4*)(pa + 16 + khalf * 8);
        }
#pragma unroll
        for (int j = 0; j < 4; ++j) {
            // B 32x16 bf16 layout: lane<16 -> K{0..7,16..23}; lane>=16 -> K{8..15,24..31}
            const bf16* pb = cB + (wn * 64 + j * 16 + lrow) * LDSS;
            bfr[j].q[0] = *(const u32x4*)(pb + khalf * 8);
            bfr[j].q[1] = *(const u32x4*)(pb + 16 + khalf * 8);
        }
#pragma unroll
        for (int r = 0; r < 2; ++r)
#pragma unroll
            for (int j = 0; j < 4; ++j)
                acc[r][j] = __builtin_amdgcn_wmma_f32_16x16x32_bf16(
                    false, af[r].v, false, bfr[j].v,
                    (short)0, acc[r][j], false, false);

        wait_asynccnt0();                // next buffer fully written (latency hidden)
        __syncthreads();                 // visible to all waves; reads of cur done
    }

    // D layout: VGPR e -> row e + 8*(lane>=16), col = lane&15
#pragma unroll
    for (int r = 0; r < 2; ++r)
#pragma unroll
        for (int j = 0; j < 4; ++j) {
            int col = n0 + wn * 64 + j * 16 + lrow;
#pragma unroll
            for (int e = 0; e < 8; ++e) {
                int row = m0 + wm * 32 + r * 16 + khalf * 8 + e;
                C[(size_t)row * N + col] = acc[r][j][e];
            }
        }
}

// ---------------------------------------------------------------------------
// Launch
// ---------------------------------------------------------------------------
extern "C" void kernel_launch(void* const* d_in, const int* in_sizes, int n_in,
                              void* d_out, int out_size, void* d_ws, size_t ws_size,
                              hipStream_t stream) {
    (void)in_sizes; (void)n_in; (void)out_size; (void)ws_size;
    const float* x   = (const float*)d_in[0];   // [8192,11,2]
    const float* bw0 = (const float*)d_in[1];   // [1024,20]
    const float* sw0 = (const float*)d_in[2];   // [1024,20,8]
    const float* sc0 = (const float*)d_in[3];   // [1024,20]
    const float* bw1 = (const float*)d_in[4];   // [1024,1024]
    const float* sw1 = (const float*)d_in[5];   // [1024,1024,8]
    const float* sc1 = (const float*)d_in[6];   // [1024,1024]
    float* out = (float*)d_out;                 // [8192,1024]

    char* ws = (char*)d_ws;
    size_t off = 0;
    auto take = [&](size_t bytes) -> char* {
        char* p = ws + off;
        off += (bytes + 255) & ~(size_t)255;
        return p;
    };
    float* roots = (float*)take((size_t)BATCH * IN1 * sizeof(float));
    bf16*  A1    = (bf16*) take((size_t)BATCH * K1PAD * sizeof(bf16));
    bf16*  W1    = (bf16*) take((size_t)HID * K1PAD * sizeof(bf16));
    float* h     = (float*)take((size_t)BATCH * HID * sizeof(float));
    bf16*  A2    = (bf16*) take((size_t)BATCH * K2 * sizeof(bf16));
    bf16*  W2    = (bf16*) take((size_t)OUTN * K2 * sizeof(bf16));

    // 1. polynomial roots (replaces companion eigvals)
    dk_roots_kernel<<<(BATCH + 255) / 256, 256, 0, stream>>>(x, roots, BATCH);

    // 2. layer 1: pack weights, expand activations, GEMM -> h
    pack_w_kernel<<<(HID * IN1 + 255) / 256, 256, 0, stream>>>(bw0, sw0, sc0, W1, HID, IN1, K1PAD);
    expand_kernel<<<(BATCH * IN1 + 255) / 256, 256, 0, stream>>>(roots, A1, BATCH, IN1, K1PAD);
    dim3 g1(HID / TILE_N, BATCH / TILE_M);
    kan_gemm_bf16<<<g1, 256, 0, stream>>>(A1, W1, h, BATCH, HID, K1PAD);

    // 3. layer 2: pack weights, expand activations, GEMM -> out
    pack_w_kernel<<<(OUTN * HID + 255) / 256, 256, 0, stream>>>(bw1, sw1, sc1, W2, OUTN, HID, K2);
    expand_kernel<<<(BATCH * HID + 255) / 256, 256, 0, stream>>>(h, A2, BATCH, HID, K2);
    dim3 g2(OUTN / TILE_N, BATCH / TILE_M);
    kan_gemm_bf16<<<g2, 256, 0, stream>>>(A2, W2, out, BATCH, OUTN, K2);
}
